// ICTDIrrepsE3ConvFullExternal_84344567759196
// MI455X (gfx1250) — matrix operand order repaired and checked by reading, not compiled, not run on hardware
//
#include <hip/hip_runtime.h>
#include <hip/hip_bf16.h>
#include <math.h>

typedef __attribute__((ext_vector_type(16))) _Float16 v16h;
typedef __attribute__((ext_vector_type(8)))  _Float16 v8h;
typedef __attribute__((ext_vector_type(8)))  float    v8f;
typedef __attribute__((ext_vector_type(4)))  float    v4f;

#define U_DIM     8
#define N_BASIS   16
#define MAX_RADIUS 5.0f
#define NPATH     15
#define FEAT_DIM  288   // 32*(1+3+5)
#define GTOT      51    // sum over paths of (2*l3+1)
#define CGTOT     615   // total CG coefficients

// Path tables (l1 outer, l2 mid, l3 inner — must match reference PATHS order)
constexpr int PK_L1[NPATH]    = {0,0,0,1,1,1,1,1,1,2,2,2,2,2,2};
constexpr int PK_L2[NPATH]    = {0,1,2,0,1,1,1,2,2,0,1,1,2,2,2};
constexpr int PK_L3[NPATH]    = {0,1,2,1,0,1,2,1,2,2,1,2,0,1,2};
constexpr int PK_GOFS[NPATH]  = {0,1,4,9,12,13,16,21,24,29,34,37,42,43,46};
constexpr int PK_CGOFS[NPATH] = {0,1,10,35,44,53,80,125,170,245,270,315,390,415,490};
// alpha = sqrt((2*l3+1)/(U*NPATH_TO[l3])); NPATH_TO = {0:3, 1:6, 2:6}
constexpr float PK_ALPHA[NPATH] = {
  0.20412414f, 0.25f, 0.32274861f, 0.25f, 0.20412414f, 0.25f, 0.32274861f,
  0.25f, 0.32274861f, 0.32274861f, 0.25f, 0.32274861f, 0.20412414f, 0.25f, 0.32274861f };
constexpr int PK_FB[NPATH] = {0,32,128,32,0,32,128,32,128,128,32,128,0,32,128};
constexpr int YOFS[3] = {0,1,4};

// ---------------------------------------------------------------------------
// CG tensor construction (double precision, single thread — negligible cost)
// ---------------------------------------------------------------------------
__device__ double dfact(int n){ double r=1.0; for(int i=2;i<=n;++i) r*=(double)i; return r; }

__device__ double su2cg(int j1,int j2,int j3,int m1,int m2,int m3){
  if (m3 != m1+m2) return 0.0;
  int vmin = -j1+j2+m3; if (-j1+m1 > vmin) vmin = -j1+m1; if (vmin < 0) vmin = 0;
  int vmax = j2+j3+m1; if (j3-j1+j2 < vmax) vmax = j3-j1+j2; if (j3+m3 < vmax) vmax = j3+m3;
  double pref = sqrt((double)(2*j3+1)*dfact(j3+j1-j2)*dfact(j3-j1+j2)*dfact(j1+j2-j3)
                     *dfact(j3+m3)*dfact(j3-m3)
                     /(dfact(j1+j2+j3+1)*dfact(j1-m1)*dfact(j1+m1)*dfact(j2-m2)*dfact(j2+m2)));
  double s = 0.0;
  for (int v=vmin; v<=vmax; ++v){
    double sg = ((v+j2+m2)&1) ? -1.0 : 1.0;
    s += sg*dfact(j2+j3+m1-v)*dfact(j1-m1+v)
         /(dfact(v)*dfact(j3-j1+j2-v)*dfact(j3+m3-v)*dfact(v+j1-j2-m3));
  }
  return pref*s;
}

__device__ void buildq(int l, double qr[5][5], double qi[5][5]){
  for (int i=0;i<5;++i) for (int j=0;j<5;++j){ qr[i][j]=0.0; qi[i][j]=0.0; }
  const double s = 0.7071067811865476;
  for (int m=-l; m<0; ++m){ qr[l+m][l-m] = s; qi[l+m][l+m] = -s; }
  qr[l][l] = 1.0;
  for (int m=1; m<=l; ++m){
    double sg = (m&1) ? -1.0 : 1.0;
    qr[l+m][l+m] = sg*s; qi[l+m][l-m] = sg*s;
  }
  // multiply by (-i)^l
  if (l == 1){
    for (int i=0;i<5;++i) for (int j=0;j<5;++j){
      double re=qr[i][j], im=qi[i][j]; qr[i][j]=im; qi[i][j]=-re;
    }
  } else if (l == 2){
    for (int i=0;i<5;++i) for (int j=0;j<5;++j){ qr[i][j]=-qr[i][j]; qi[i][j]=-qi[i][j]; }
  }
}

__global__ void cg_init_kernel(float* __restrict__ cg){
  if (threadIdx.x != 0 || blockIdx.x != 0) return;
  for (int p=0; p<NPATH; ++p){
    const int l1=PK_L1[p], l2=PK_L2[p], l3=PK_L3[p];
    const int d1=2*l1+1, d2=2*l2+1, d3=2*l3+1;
    double q1r[5][5],q1i[5][5],q2r[5][5],q2i[5][5],q3r[5][5],q3i[5][5];
    buildq(l1,q1r,q1i); buildq(l2,q2r,q2i); buildq(l3,q3r,q3i);
    double C[5][5][5]; double nrm = 0.0;
    for (int jo=0;jo<d1;++jo) for (int lo=0;lo<d2;++lo) for (int mo=0;mo<d3;++mo){
      double sre = 0.0;
      for (int ii=0;ii<d1;++ii) for (int kk=0;kk<d2;++kk) for (int nn=0;nn<d3;++nn){
        double cc = su2cg(l1,l2,l3, ii-l1, kk-l2, nn-l3);
        if (cc == 0.0) continue;
        double tre = q1r[ii][jo]*q2r[kk][lo] - q1i[ii][jo]*q2i[kk][lo];
        double tim = q1r[ii][jo]*q2i[kk][lo] + q1i[ii][jo]*q2r[kk][lo];
        sre += (tre*q3r[nn][mo] + tim*q3i[nn][mo]) * cc;
      }
      C[jo][lo][mo] = sre; nrm += sre*sre;
    }
    double inv = 1.0 / sqrt(nrm);
    for (int jo=0;jo<d1;++jo) for (int lo=0;lo<d2;++lo) for (int mo=0;mo<d3;++mo)
      cg[PK_CGOFS[p] + (jo*d2+lo)*d3 + mo] = (float)(C[jo][lo][mo]*inv);
  }
}

// ---------------------------------------------------------------------------
// Small helpers / aux kernels
// ---------------------------------------------------------------------------
__device__ __forceinline__ float fast_rcp(float x){ return __builtin_amdgcn_rcpf(x); }
__device__ __forceinline__ float siluf(float x){ return x * fast_rcp(1.0f + __expf(-x)); }

__global__ void zero_kernel(float* __restrict__ p, int n){
  int i = blockIdx.x*256 + threadIdx.x;
  if (i < n) p[i] = 0.0f;
}

__global__ void cvt_f16_kernel(const float* __restrict__ src, _Float16* __restrict__ dst, int n){
  int i = blockIdx.x*256 + threadIdx.x;
  if (i < n) dst[i] = (_Float16)src[i];
}

__global__ void node_mlp_kernel(const int* __restrict__ A, const float* __restrict__ embt,
                                const float* __restrict__ w1, const float* __restrict__ b1,
                                const float* __restrict__ w2, const float* __restrict__ b2,
                                float* __restrict__ Ai, int N){
  int n = blockIdx.x*blockDim.x + threadIdx.x;
  if (n >= N) return;
  const float* e = embt + A[n]*16;
  float ein[16];
#pragma unroll
  for (int i=0;i<16;++i) ein[i] = e[i];
  float h[64];
#pragma unroll
  for (int j=0;j<64;++j){
    float s = b1[j];
#pragma unroll
    for (int i=0;i<16;++i) s += ein[i]*w1[i*64+j];
    h[j] = siluf(s);
  }
#pragma unroll
  for (int u=0;u<U_DIM;++u){
    float s = b2[u];
#pragma unroll
    for (int j=0;j<64;++j) s += h[j]*w2[j*U_DIM+u];
    Ai[n*U_DIM+u] = s;
  }
}

// ---------------------------------------------------------------------------
// Fused edge kernel: geometry -> radial MLP (WMMA) -> big GEMM (WMMA) ->
// tensor-product fold -> atomic scatter to nodes
// ---------------------------------------------------------------------------
__device__ __forceinline__ v8f wmma_f16(v16h a, v16h b, v8f c){
  return __builtin_amdgcn_wmma_f32_16x16x32_f16(false, a, false, b, (short)0, c, false, false);
}

// A-fragment (16xK, f16, row-major source with given stride) per ISA layout:
// lane m = lane&15, hi = lane>>4; elems 0..7 = K[k0+8*hi .. +7], 8..15 = K[k0+16+8*hi ..]
__device__ __forceinline__ v16h load_a_frag(const _Float16* base, int stride, int k0, int lane){
  const int m = lane & 15, hi = lane >> 4;
  const v8h lo = *reinterpret_cast<const v8h*>(base + m*stride + k0 + 8*hi);
  const v8h hp = *reinterpret_cast<const v8h*>(base + m*stride + k0 + 16 + 8*hi);
  v16h r;
#pragma unroll
  for (int i=0;i<8;++i){ r[i] = lo[i]; r[i+8] = hp[i]; }
  return r;
}

__device__ __forceinline__ v16h load_b_frag(const _Float16* base, int stride, int k0, int n0, int lane){
  const int n = lane & 15, hi = lane >> 4;
  v16h r;
#pragma unroll
  for (int i=0;i<8;++i){
    r[i]   = base[(k0 + 8*hi + i)*stride + n0 + n];
    r[i+8] = base[(k0 + 16 + 8*hi + i)*stride + n0 + n];
  }
  return r;
}

#define WPB 2
#define TPB (WPB*32)

struct alignas(32) WaveArena {
  _Float16 emb[16*32];       // radial basis, padded K 16->32
  _Float16 h[16*64];         // h1 then h2 (f16 A-operand staging)
  float    Ae[U_DIM*16];     // gathered node weights, TRANSPOSED: [u][m]
  float    G[16*GTOT];       // per-path Y(x)Y CG contraction
  float    feat[16*FEAT_DIM];// per-edge output features
};

__launch_bounds__(TPB, 1)
__global__ void edge_tp_kernel(
    const float* __restrict__ pos,    const int* __restrict__ batch,
    const int* __restrict__ esrc,     const int* __restrict__ edst,
    const float* __restrict__ shifts, const float* __restrict__ cell,
    const float* __restrict__ Ai,
    const float* __restrict__ w_fc1,  const float* __restrict__ b_fc1,
    const float* __restrict__ w_fc2,  const float* __restrict__ b_fc2,
    const _Float16* __restrict__ w3h, const float* __restrict__ b_fc3,
    const float* __restrict__ cgbuf,
    float* __restrict__ out,
    int E, float scale)
{
  __shared__ alignas(32) _Float16 sW1[32*64];  // w_fc1 padded K 16->32
  __shared__ alignas(32) _Float16 sW2[64*64];
  __shared__ float sB1[64], sB2[64];
  __shared__ WaveArena arena[WPB];

  const int tid = threadIdx.x;
  for (int i = tid; i < 32*64; i += TPB) sW1[i] = (i < 16*64) ? (_Float16)w_fc1[i] : (_Float16)0.0f;
  for (int i = tid; i < 64*64; i += TPB) sW2[i] = (_Float16)w_fc2[i];
  if (tid < 64){ sB1[tid] = b_fc1[tid]; sB2[tid] = b_fc2[tid]; }
  __syncthreads();

  const int wv   = tid >> 5;
  const int lane = tid & 31;
  const int m16  = lane & 15;
  const int hi   = lane >> 4;
  WaveArena& ar = arena[wv];

  const int tile = blockIdx.x * WPB + wv;
  const int base = tile * 16;
  if (base >= E) return;

  for (int i = lane; i < 16*FEAT_DIM; i += 32) ar.feat[i] = 0.0f;

  // ---- Phase A: one lane per edge: geometry, Y, G, Ae gather, radial basis ----
  if (lane < 16){
    const int e = base + lane;
    float Y[9];
#pragma unroll
    for (int i=0;i<9;++i) Y[i] = 0.0f;
    if (e < E){
      const int s  = esrc[e];
      const int d  = edst[e];
      const int bg = batch[s];
      const float sh0 = shifts[e*3+0], sh1 = shifts[e*3+1], sh2 = shifts[e*3+2];
      const float* cl = cell + bg*9;
      const float vx = pos[d*3+0]-pos[s*3+0] + sh0*cl[0]+sh1*cl[3]+sh2*cl[6];
      const float vy = pos[d*3+1]-pos[s*3+1] + sh0*cl[1]+sh1*cl[4]+sh2*cl[7];
      const float vz = pos[d*3+2]-pos[s*3+2] + sh0*cl[2]+sh1*cl[5]+sh2*cl[8];
      const float len = sqrtf(vx*vx+vy*vy+vz*vz);
      const float inv = fast_rcp(fmaxf(len, 1e-8f));
      const float x = vx*inv, y = vy*inv, z = vz*inv;
      const float s3c = 1.7320508075688772f, s5c = 2.2360679774997896f, s15c = 3.872983346207417f;
      Y[0]=1.0f; Y[1]=s3c*y; Y[2]=s3c*z; Y[3]=s3c*x;
      Y[4]=s15c*x*y; Y[5]=s15c*y*z; Y[6]=0.5f*s5c*(3.0f*z*z-1.0f);
      Y[7]=s15c*x*z; Y[8]=0.5f*s15c*(x*x-y*y);
#pragma unroll
      for (int u=0;u<U_DIM;++u) ar.Ae[u*16+lane] = Ai[s*U_DIM+u];   // transposed [u][m]
      const float stepinv = 17.0f / MAX_RADIUS;
#pragma unroll
      for (int i=0;i<N_BASIS;++i){
        const float v  = (float)(i+1) * (MAX_RADIUS/17.0f);
        const float df = (len - v) * stepinv;
        ar.emb[lane*32+i] = (_Float16)(__expf(-df*df) * (4.0f/1.12f));
      }
    } else {
#pragma unroll
      for (int u=0;u<U_DIM;++u) ar.Ae[u*16+lane] = 0.0f;
#pragma unroll
      for (int i=0;i<N_BASIS;++i) ar.emb[lane*32+i] = (_Float16)0.0f;
    }
#pragma unroll
    for (int i=N_BASIS;i<32;++i) ar.emb[lane*32+i] = (_Float16)0.0f;

    // G[p][k] = sum_{a,b} Y_l1[a]*Y_l2[b]*CG[a,b,k]   (fully unrolled, uniform cg loads)
#pragma unroll
    for (int p=0;p<NPATH;++p){
      const int l1=PK_L1[p], l2=PK_L2[p], l3=PK_L3[p];
      const int d1=2*l1+1, d2=2*l2+1, d3=2*l3+1;
#pragma unroll
      for (int k=0;k<d3;++k){
        float g = 0.0f;
#pragma unroll
        for (int a=0;a<d1;++a)
#pragma unroll
          for (int b=0;b<d2;++b)
            g += Y[YOFS[l1]+a]*Y[YOFS[l2]+b]*cgbuf[PK_CGOFS[p] + (a*d2+b)*d3 + k];
        ar.G[lane*GTOT + PK_GOFS[p] + k] = g;
      }
    }
  }

  // ---- GEMM1: h1 = silu(emb @ W1 + b1)  [16x32 @ 32x64] ----
  {
    const v16h a = load_a_frag(ar.emb, 32, 0, lane);
#pragma unroll
    for (int t=0;t<4;++t){
      const v16h b = load_b_frag(sW1, 64, 0, t*16, lane);
      v8f acc = {};
      acc = wmma_f16(a, b, acc);
      const float bias = sB1[t*16 + m16];
#pragma unroll
      for (int r=0;r<8;++r)
        ar.h[(hi*8+r)*64 + t*16 + m16] = (_Float16)siluf(acc[r] + bias);
    }
  }

  // ---- GEMM2: h2 = silu(h1 @ W2 + b2)  [16x64 @ 64x64] ----
  {
    const v16h a0 = load_a_frag(ar.h, 64, 0, lane);
    const v16h a1 = load_a_frag(ar.h, 64, 32, lane);
#pragma unroll
    for (int t=0;t<4;++t){
      const v16h b0 = load_b_frag(sW2, 64, 0, t*16, lane);
      const v16h b1 = load_b_frag(sW2, 64, 32, t*16, lane);
      v8f acc = {};
      acc = wmma_f16(a0, b0, acc);
      acc = wmma_f16(a1, b1, acc);
      const float bias = sB2[t*16 + m16];
#pragma unroll
      for (int r=0;r<8;++r)
        ar.h[(hi*8+r)*64 + t*16 + m16] = (_Float16)siluf(acc[r] + bias);
    }
  }

  // ---- Main loop: [16x64] @ w_fc3[64 x 3840] in 16-col tiles, folded on the fly.
  //      Path loop unrolled (per-path constants become immediates); the u loop
  //      stays ROLLED to bound code size and register pressure. ----
  const v16h a0 = load_a_frag(ar.h, 64, 0, lane);
  const v16h a1 = load_a_frag(ar.h, 64, 32, lane);
  const _Float16* wrow0 = w3h + lane*3840;
  const _Float16* wrow1 = w3h + (lane+32)*3840;

#pragma unroll
  for (int p=0;p<NPATH;++p){
    float tAcc[16];
#pragma unroll
    for (int i=0;i<16;++i) tAcc[i] = 0.0f;

    const int pbase = p*256;
    // preload u=0 (both halves)
    v16h nb00 = load_b_frag(w3h, 3840, 0, pbase, lane);
    v16h nb10 = load_b_frag(w3h, 3840, 32, pbase, lane);
    v16h nb01 = load_b_frag(w3h, 3840, 0, pbase + 16, lane);
    v16h nb11 = load_b_frag(w3h, 3840, 32, pbase + 16, lane);
    float nbi0 = b_fc3[pbase + m16];
    float nbi1 = b_fc3[pbase + 16 + m16];

#pragma unroll 1
    for (int u=0;u<U_DIM;++u){
      // Ae column for this u: contiguous 8 floats per lane (transposed layout)
      const v4f* aep = reinterpret_cast<const v4f*>(ar.Ae + u*16 + hi*8);
      const v4f ae0 = aep[0], ae1 = aep[1];
      float Aev[8];
#pragma unroll
      for (int r=0;r<4;++r){ Aev[r] = ae0[r]; Aev[r+4] = ae1[r]; }

      const v16h b00 = nb00, b10 = nb10, b01 = nb01, b11 = nb11;
      const float bi0 = nbi0, bi1 = nbi1;
      if (u+1 < U_DIM){
        const int off = pbase + (u+1)*32;
        nb00 = load_b_frag(w3h, 3840, 0, off, lane);
        nb10 = load_b_frag(w3h, 3840, 32, off, lane);
        nb01 = load_b_frag(w3h, 3840, 0, off + 16, lane);
        nb11 = load_b_frag(w3h, 3840, 32, off + 16, lane);
        nbi0 = b_fc3[off + m16];
        nbi1 = b_fc3[off + 16 + m16];
      }

      v8f acc = {};
      acc = wmma_f16(a0, b00, acc);
      acc = wmma_f16(a1, b10, acc);
#pragma unroll
      for (int r=0;r<8;++r) tAcc[r*2+0] += Aev[r] * (acc[r] + bi0);

      v8f acc2 = {};
      acc2 = wmma_f16(a0, b01, acc2);
      acc2 = wmma_f16(a1, b11, acc2);
#pragma unroll
      for (int r=0;r<8;++r) tAcc[r*2+1] += Aev[r] * (acc2[r] + bi1);
    }

    // prefetch next path's first tile while we do the fold
    if (p+1 < NPATH){
      __builtin_prefetch(wrow0 + (p+1)*256, 0, 0);
      __builtin_prefetch(wrow1 + (p+1)*256, 0, 0);
    }

    // fold path result with G into feat (no cross-lane collisions -> plain RMW)
    const int d3 = 2*PK_L3[p]+1;
    const float alpha = PK_ALPHA[p];
    const int go = PK_GOFS[p];
    const int fb = PK_FB[p];
#pragma unroll
    for (int k=0;k<d3;++k){
#pragma unroll
      for (int r=0;r<8;++r){
        const int m = hi*8 + r;
        const float g = alpha * ar.G[m*GTOT + go + k];
        ar.feat[m*FEAT_DIM + fb + m16*d3 + k]      += g * tAcc[r*2+0];
        ar.feat[m*FEAT_DIM + fb + (m16+16)*d3 + k] += g * tAcc[r*2+1];
      }
    }
  }

  // ---- scatter: node_out[dst] += feat * (num_nodes/E) ----
#pragma unroll 1
  for (int m=0;m<16;++m){
    const int e = base + m;
    if (e >= E) continue;
    const int dn = edst[e];
    float* op = out + dn*FEAT_DIM;
    const float* fp = ar.feat + m*FEAT_DIM;
    for (int idx = lane; idx < FEAT_DIM; idx += 32){
      __hip_atomic_fetch_add(op + idx, fp[idx]*scale,
                             __ATOMIC_RELAXED, __HIP_MEMORY_SCOPE_AGENT);
    }
  }
}

// ---------------------------------------------------------------------------
extern "C" void kernel_launch(void* const* d_in, const int* in_sizes, int n_in,
                              void* d_out, int out_size, void* d_ws, size_t ws_size,
                              hipStream_t stream)
{
  const float* pos    = (const float*)d_in[0];
  const int*   A      = (const int*)  d_in[1];
  const int*   batch  = (const int*)  d_in[2];
  const int*   esrc   = (const int*)  d_in[3];
  const int*   edst   = (const int*)  d_in[4];
  const float* shifts = (const float*)d_in[5];
  const float* cell   = (const float*)d_in[6];
  const float* embt   = (const float*)d_in[7];
  const float* w_am1  = (const float*)d_in[8];
  const float* b_am1  = (const float*)d_in[9];
  const float* w_am2  = (const float*)d_in[10];
  const float* b_am2  = (const float*)d_in[11];
  const float* w_fc1  = (const float*)d_in[12];
  const float* b_fc1  = (const float*)d_in[13];
  const float* w_fc2  = (const float*)d_in[14];
  const float* b_fc2  = (const float*)d_in[15];
  const float* w_fc3  = (const float*)d_in[16];
  const float* b_fc3  = (const float*)d_in[17];
  float* out = (float*)d_out;

  const int N = in_sizes[0] / 3;
  const int E = in_sizes[3];

  char* ws = (char*)d_ws;
  float* cgbuf = (float*)ws;                       // CGTOT floats
  float* Ai    = (float*)(ws + 4096);              // N*8 floats
  size_t aiBytes = (((size_t)N*U_DIM*4) + 255) & ~(size_t)255;
  _Float16* w3h = (_Float16*)(ws + 4096 + aiBytes); // 64*3840 halves (f16 copy of w_fc3)

  cg_init_kernel<<<1, 1, 0, stream>>>(cgbuf);
  zero_kernel<<<(out_size + 255)/256, 256, 0, stream>>>(out, out_size);
  node_mlp_kernel<<<(N + 127)/128, 128, 0, stream>>>(A, embt, w_am1, b_am1, w_am2, b_am2, Ai, N);
  {
    const int n = 64*3840;
    cvt_f16_kernel<<<(n + 255)/256, 256, 0, stream>>>(w_fc3, w3h, n);
  }
  const float avg = (float)E / (float)N;
  const float scale = 1.0f / ((avg > 1e-8f) ? avg : 1e-8f);
  const int tiles = (E + 15)/16;
  const int blocks = (tiles + WPB - 1)/WPB;
  edge_tp_kernel<<<blocks, TPB, 0, stream>>>(pos, batch, esrc, edst, shifts, cell,
      Ai, w_fc1, b_fc1, w_fc2, b_fc2, w3h, b_fc3, cgbuf, out, E, scale);
}
